// RandomWalkMetropolisHastings_75943611728568
// MI455X (gfx1250) — matrix operand order repaired and checked by reading, not compiled, not run on hardware
//
#include <hip/hip_runtime.h>
#include <hip/hip_bf16.h>
#include <cstdint>

typedef __attribute__((ext_vector_type(16))) _Float16 v16h;
typedef __attribute__((ext_vector_type(8)))  float    v8f;

#define DIM   64      // walker dimension
#define HID   128     // hidden width
#define WPB   64      // walkers per block
#define TPB   128     // 4 waves of 32
#define PSTD  0.3f    // proposal std

// ---- stateless counter RNG (deterministic across replays) ----
__device__ __forceinline__ uint32_t hash32(uint32_t x) {
  x ^= x >> 16; x *= 0x7feb352dU;
  x ^= x >> 15; x *= 0x846ca68bU;
  x ^= x >> 16; return x;
}
__device__ __forceinline__ float u01(uint32_t h) {
  return ((h >> 8) + 0.5f) * (1.0f / 16777216.0f);
}

// gfx1250 v_tanh_f32 if the toolchain exposes it; branch-free exp fallback.
__device__ __forceinline__ float fast_tanh(float x) {
#if __has_builtin(__builtin_amdgcn_tanhf)
  return __builtin_amdgcn_tanhf(x);
#else
  const float e = __expf(2.0f * x);          // v_exp_f32
  return 1.0f - 2.0f / (e + 1.0f);           // v_rcp_f32, no branches
#endif
}

__global__ __launch_bounds__(TPB)
void RWMH_wmma_kernel(const float* __restrict__ chains,
                      const float* __restrict__ W1,
                      const float* __restrict__ b1,
                      const float* __restrict__ W2,
                      const float* __restrict__ b2,
                      const int* __restrict__ p_ns,
                      const int* __restrict__ p_burn,
                      const int* __restrict__ p_thin,
                      float* __restrict__ out,
                      int nwalkers)
{
  __shared__ _Float16 sW1[DIM * HID];                    // 16 KB
  __shared__ __align__(16) float sChain[WPB * DIM];      // 16 KB
  __shared__ __align__(16) float sCand [WPB * DIM];      // 16 KB
  __shared__ float sB1[HID];
  __shared__ float sW2[HID];
  __shared__ float sLogCur [WPB];
  __shared__ float sLogCand[WPB];
  __shared__ int   sFlag[WPB];
  __shared__ float sAcc [WPB];
  __shared__ float sB2;

  const int tid    = threadIdx.x;
  const int lane   = tid & 31;
  const int wave   = tid >> 5;
  const int l15    = lane & 15;
  const int hi     = (lane >> 4) & 1;       // 0: lanes 0-15, 1: lanes 16-31
  const int wgBase = blockIdx.x * WPB;      // first global walker of this block

  const int ns    = p_ns[0];
  const int burn  = p_burn[0];
  const int thin  = p_thin[0];
  const int total = thin * (ns + 1) + burn; // 578 for the reference config

  // Nudge the tiny weight arrays toward L2 (global_prefetch_b8 path).
  __builtin_prefetch(W1 + tid * 16, 0, 1);

  // ---- stage weights + chain state into LDS ----
  for (int i = tid; i < DIM * HID; i += TPB) sW1[i] = (_Float16)W1[i];
  for (int i = tid; i < HID; i += TPB) { sB1[i] = b1[i]; sW2[i] = W2[i]; }
  if (tid == 0) sB2 = b2[0];
  for (int i = tid; i < WPB * DIM; i += TPB)
    sChain[i] = chains[(size_t)wgBase * DIM + i];
  for (int i = tid; i < WPB; i += TPB) sAcc[i] = 0.0f;
  __syncthreads();

  // ---- hoist W1 into WMMA B-fragments: 2 K-chunks x 8 N-tiles, kept in VGPRs
  // B (32x16, KxN) layout: lane = N (mod 16), lanes 16-31 offset K by 16,
  // element e -> K = chunk*32 + hi*16 + e.
  v16h bfrag[2][8];
  #pragma unroll
  for (int c = 0; c < 2; ++c)
    #pragma unroll
    for (int t = 0; t < 8; ++t) {
      v16h b;
      #pragma unroll
      for (int e = 0; e < 16; ++e) {
        const int k = c * 32 + hi * 16 + e;
        const int n = t * 16 + l15;
        b[e] = sW1[k * HID + n];
      }
      bfrag[c][t] = b;
    }

  // A-fragment builder: row is contiguous in LDS -> ds_load_b128s + pk cvt.
  // A (16x32 f16) layout: element e -> K = c*32 + hi*16 + e.
  auto mkA = [&](const float* __restrict__ src, int row, int c) -> v16h {
    const float4* rp = (const float4*)(src + row * DIM);
    const float4 q0 = rp[c * 8 + hi * 4];       // K = c*32 + hi*16 + 0..3
    const float4 q1 = rp[c * 8 + hi * 4 + 1];   //                + 4..7
    const float4 q2 = rp[c * 8 + hi * 4 + 2];   //                + 8..11
    const float4 q3 = rp[c * 8 + hi * 4 + 3];   //                + 12..15
    v16h a;
    a[0]  = (_Float16)q0.x; a[1]  = (_Float16)q0.y;
    a[2]  = (_Float16)q0.z; a[3]  = (_Float16)q0.w;
    a[4]  = (_Float16)q1.x; a[5]  = (_Float16)q1.y;
    a[6]  = (_Float16)q1.z; a[7]  = (_Float16)q1.w;
    a[8]  = (_Float16)q2.x; a[9]  = (_Float16)q2.y;
    a[10] = (_Float16)q2.z; a[11] = (_Float16)q2.w;
    a[12] = (_Float16)q3.x; a[13] = (_Float16)q3.y;
    a[14] = (_Float16)q3.z; a[15] = (_Float16)q3.w;
    return a;
  };

  // ---- fused MLP log-pdf for this wave's 16 walkers:
  //      dst[w] = 2*(tanh(x@W1 + b1) @ W2 + b2)
  auto logpdf = [&](const float* __restrict__ src, float* __restrict__ dst) {
    const int row = wave * 16 + l15;            // A layout: lane -> M = lane&15
    const v16h a0 = mkA(src, row, 0);
    const v16h a1 = mkA(src, row, 1);

    float part[8];
    #pragma unroll
    for (int r = 0; r < 8; ++r) part[r] = 0.0f;

    #pragma unroll
    for (int t = 0; t < 8; ++t) {
      v8f acc = {};
      acc = __builtin_amdgcn_wmma_f32_16x16x32_f16(false, a0, false, bfrag[0][t],
                                                   (short)0, acc, false, false);
      acc = __builtin_amdgcn_wmma_f32_16x16x32_f16(false, a1, false, bfrag[1][t],
                                                   (short)0, acc, false, false);
      // C layout: VGPR r -> row r (lanes 0-15) / r+8 (lanes 16-31), N = lane&15
      const int   n    = t * 16 + l15;
      const float bias = sB1[n];
      const float w2v  = sW2[n];
      #pragma unroll
      for (int r = 0; r < 8; ++r)
        part[r] += fast_tanh(acc[r] + bias) * w2v;
    }
    // reduce the 16 lanes of each half-wave (masks 1,2,4,8 stay inside a half)
    #pragma unroll
    for (int m = 1; m <= 8; m <<= 1)
      #pragma unroll
      for (int r = 0; r < 8; ++r)
        part[r] += __shfl_xor(part[r], m, 32);
    if (l15 == 0) {
      #pragma unroll
      for (int r = 0; r < 8; ++r)
        dst[wave * 16 + hi * 8 + r] = 2.0f * (part[r] + sB2);
    }
  };

  logpdf(sChain, sLogCur);   // initial cached log-pdf
  __syncthreads();

  for (int step = 0; step < total; ++step) {
    // ---- propose: xcand = x + 0.3*N(0,1); full Box-Muller pair per thread
    for (int i = tid; i < (WPB * DIM) / 2; i += TPB) {
      const uint32_t s = (uint32_t)step * (uint32_t)(nwalkers * DIM / 2)
                       + (uint32_t)(wgBase * (DIM / 2) + i);
      const float u1 = u01(hash32(s * 2654435761u + 0x9e3779b9u));
      const float u2 = u01(hash32(s * 2246822519u + 0x85ebca6bu));
      const float r  = sqrtf(-2.0f * __logf(u1));
      float sn, cs;
      __sincosf(6.28318530717958647f * u2, &sn, &cs);
      sCand[2 * i]     = sChain[2 * i]     + PSTD * r * cs;
      sCand[2 * i + 1] = sChain[2 * i + 1] + PSTD * r * sn;
    }
    __syncthreads();

    logpdf(sCand, sLogCand);
    __syncthreads();

    // ---- accept/reject (one thread per walker)
    if (tid < WPB) {
      const uint32_t s = hash32(0xC0FFEEu ^
                                ((uint32_t)step * (uint32_t)nwalkers +
                                 (uint32_t)(wgBase + tid)));
      const float log_u = __logf(u01(s));
      const int acc = (log_u < (sLogCand[tid] - sLogCur[tid])) ? 1 : 0;
      sFlag[tid] = acc;
      if (acc) { sLogCur[tid] = sLogCand[tid]; sAcc[tid] += 1.0f; }
    }
    __syncthreads();

    {
      const float4* c4 = (const float4*)sCand;
      float4*       h4 = (float4*)sChain;
      for (int i = tid; i < (WPB * DIM) / 4; i += TPB)
        if (sFlag[i >> 4]) h4[i] = c4[i];     // 16 float4 per walker row
    }
    __syncthreads();

    // ---- thinning write-out: values[walker, slot, :] via async LDS->global
    if (step > burn && (step % thin) == 0) {
      const int slot = (step - burn) / thin - 1;
      const int wl   = tid >> 1;              // walker local (0..63)
      const int half = tid & 1;               // 32 contiguous floats per thread
      float* gbase = out + ((size_t)(wgBase + wl) * ns + slot) * DIM + half * 32;
      const float* lbase = sChain + wl * DIM + half * 32;
      #pragma unroll
      for (int q = 0; q < 8; ++q) {
        float*   gp   = gbase + q * 4;
        uint32_t lofs = (uint32_t)(uintptr_t)(lbase + q * 4);  // LDS byte addr
        // GLOBAL_STORE_ASYNC_FROM_LDS_B128, GV mode (saddr=off), ASYNCcnt
        asm volatile("global_store_async_from_lds_b128 %0, %1, off"
                     :: "v"(gp), "v"(lofs) : "memory");
      }
      asm volatile("s_wait_asynccnt 0x0" ::: "memory");
    }
    // next write to sChain is 3 barriers away -> no extra barrier needed here
  }
  __syncthreads();

  // ---- acceptance percentage, appended after values
  if (tid < WPB)
    out[(size_t)nwalkers * (size_t)ns * DIM + wgBase + tid] =
        100.0f * sAcc[tid] / (float)total;
}

extern "C" void kernel_launch(void* const* d_in, const int* in_sizes, int n_in,
                              void* d_out, int out_size, void* d_ws, size_t ws_size,
                              hipStream_t stream) {
  const float* chains = (const float*)d_in[0];
  const float* W1     = (const float*)d_in[1];
  const float* b1     = (const float*)d_in[2];
  const float* W2     = (const float*)d_in[3];
  const float* b2     = (const float*)d_in[4];
  const int*   ns     = (const int*)d_in[5];
  const int*   burn   = (const int*)d_in[6];
  const int*   thin   = (const int*)d_in[7];

  const int nwalkers = in_sizes[0] / DIM;   // 2048
  const int nblocks  = nwalkers / WPB;      // 32 independent workgroups

  RWMH_wmma_kernel<<<nblocks, TPB, 0, stream>>>(
      chains, W1, b1, W2, b2, ns, burn, thin, (float*)d_out, nwalkers);
}
